// RNN_16449724744958
// MI455X (gfx1250) — matrix operand order repaired and checked by reading, not compiled, not run on hardware
//
#include <hip/hip_runtime.h>

// LSTM: B=8192, T=168, F=64, H=50, HEAD=100
// z = x_t@Wx + h@Wh + b ; gates -> c,h ; out = relu(h@W1+b1)@W2+b2 + x[:,-1,0:1]

#define B_    8192
#define T_    168
#define F_    64
#define H_    50
#define HEAD_ 100
#define GATES 200   // 4*H
#define KPAD  64    // padded K for h@Wh (and hbuf width)

typedef __attribute__((ext_vector_type(16))) _Float16 v16h;
typedef __attribute__((ext_vector_type(8)))  float    v8f;

__device__ __forceinline__ float sigmoidf_(float x) {
    return 1.0f / (1.0f + __expf(-x));               // stable for all x
}
__device__ __forceinline__ float tanhf_(float x) {
    return 1.0f - 2.0f / (__expf(2.0f * x) + 1.0f);  // stable: ->±1 at extremes
}

// A-frag k offset (16-bit A, 16x32): half 2v+p holds A[m][k],
// m = laneN, k = kb*32 + laneHalf*8 + (v<4 ? 2v : 16+2(v-4)) + p
__device__ __forceinline__ int a_k0(int kb, int laneHalf, int v) {
    return kb * 32 + laneHalf * 8 + ((v < 4) ? 2 * v : 16 + 2 * (v - 4));
}

__global__ __launch_bounds__(128, 1)
void lstm_wmma_kernel(const float* __restrict__ x,
                      const float* __restrict__ Wx,
                      const float* __restrict__ Wh,
                      const float* __restrict__ b,
                      const float* __restrict__ W1,
                      const float* __restrict__ b1,
                      const float* __restrict__ W2,
                      const float* __restrict__ b2,
                      float* __restrict__ out) {
    // LDS: z gate buffer [col(256)][row(16)] f32, h state [row(16)][k(64)] f16,
    //      c state [j(50)][row(16)] f32, final-h fp32 copy, head row accumulators.
    __shared__ float    zbuf[256 * 16];
    __shared__ _Float16 hbuf[16 * KPAD];
    __shared__ float    cbuf[H_ * 16];
    __shared__ float    hf[16 * H_];
    __shared__ float    rowacc[16];

    const int tid      = threadIdx.x;
    const int lane     = tid & 31;
    const int wave     = tid >> 5;       // 0..3, owns N-tiles wave*4 .. wave*4+3
    const int laneN    = lane & 15;      // m index (A) and n index (B/C/D)
    const int laneHalf = lane >> 4;      // 0/1
    const int tile     = blockIdx.x;     // 16 batch rows per block

    // ---- zero the recurrent state (h=0, c=0; hbuf padded cols too) ----
    for (int i = tid; i < 16 * KPAD; i += 128) hbuf[i] = (_Float16)0.0f;
    for (int i = tid; i < H_ * 16;  i += 128) cbuf[i] = 0.0f;

    // ---- load packed weight B-fragments into registers (reused 168x) ----
    // B frag (16-bit, 32x16): half 2v+p holds B[k][n], k = kb*32 + laneHalf*8 + (v<4 ? 2v : 16+2(v-4)) + p,
    // n = nt*16 + laneN. Padded column n -> gate g = n>>6, unit j = n&63 (j<50 valid).
    v16h wxf[2][4];
    v16h whf[2][4];
    float bcol[4];
    #pragma unroll
    for (int i = 0; i < 4; ++i) {
        const int nt = wave * 4 + i;
        const int n  = nt * 16 + laneN;
        const int g  = n >> 6;
        const int j  = n & 63;
        bcol[i] = (j < H_) ? b[g * H_ + j] : 0.0f;
        #pragma unroll
        for (int kb = 0; kb < 2; ++kb) {
            #pragma unroll
            for (int v = 0; v < 8; ++v) {
                #pragma unroll
                for (int p = 0; p < 2; ++p) {
                    const int k = a_k0(kb, laneHalf, v) + p;
                    float wx = 0.0f, wh = 0.0f;
                    if (j < H_) {
                        wx = Wx[k * GATES + g * H_ + j];
                        if (k < H_) wh = Wh[k * GATES + g * H_ + j];
                    }
                    wxf[kb][i][2 * v + p] = (_Float16)wx;
                    whf[kb][i][2 * v + p] = (_Float16)wh;
                }
            }
        }
    }

    __syncthreads();

    const float* xrow = x + ((size_t)(tile * 16 + laneN)) * T_ * F_;

    // ---- software pipeline: x tile for step t+1 is loaded AFTER the first
    //      barrier of step t, so its latency is hidden by the elementwise
    //      phase + second barrier, and the pre-barrier wait only drains DScnt. ----
    float2 xraw[2][8];
    #pragma unroll
    for (int kb = 0; kb < 2; ++kb)
        #pragma unroll
        for (int v = 0; v < 8; ++v)
            xraw[kb][v] = *(const float2*)(xrow + a_k0(kb, laneHalf, v));

    for (int t = 0; t < T_; ++t) {
        // convert previously-loaded x registers to f16 A-fragments
        v16h ax[2];
        #pragma unroll
        for (int kb = 0; kb < 2; ++kb)
            #pragma unroll
            for (int v = 0; v < 8; ++v) {
                ax[kb][2 * v]     = (_Float16)xraw[kb][v].x;
                ax[kb][2 * v + 1] = (_Float16)xraw[kb][v].y;
            }

        // h A-fragments from LDS, packed 2xf16 -> merged ds_load_b128
        union AH { v16h v; unsigned int u[8]; } ah[2];
        #pragma unroll
        for (int kb = 0; kb < 2; ++kb)
            #pragma unroll
            for (int v = 0; v < 8; ++v)
                ah[kb].u[v] = *(const unsigned int*)&hbuf[laneN * KPAD + a_k0(kb, laneHalf, v)];

        // z = b + x@Wx + h@Wh for this wave's 4 N-tiles (4 chained WMMAs each)
        #pragma unroll
        for (int i = 0; i < 4; ++i) {
            v8f acc;
            #pragma unroll
            for (int r = 0; r < 8; ++r) acc[r] = bcol[i];
            acc = __builtin_amdgcn_wmma_f32_16x16x32_f16(false, ax[0],   false, wxf[0][i], (short)0, acc, false, false);
            acc = __builtin_amdgcn_wmma_f32_16x16x32_f16(false, ax[1],   false, wxf[1][i], (short)0, acc, false, false);
            acc = __builtin_amdgcn_wmma_f32_16x16x32_f16(false, ah[0].v, false, whf[0][i], (short)0, acc, false, false);
            acc = __builtin_amdgcn_wmma_f32_16x16x32_f16(false, ah[1].v, false, whf[1][i], (short)0, acc, false, false);
            // C/D layout: acc[r] = D[m][n], m = laneHalf*8 + r, n = nt*16 + laneN
            const int n = (wave * 4 + i) * 16 + laneN;
            float* zp = &zbuf[n * 16 + laneHalf * 8];
            #pragma unroll
            for (int r = 0; r < 8; ++r) zp[r] = acc[r];
        }
        __syncthreads();   // zbuf visible; hbuf reads above done before rewrite below

        // issue next step's x loads now (post-barrier): they complete during
        // the elementwise phase; no load-wait sits in front of the barrier.
        if (t + 1 < T_) {
            const float* xn = xrow + (size_t)(t + 1) * F_;
            #pragma unroll
            for (int kb = 0; kb < 2; ++kb)
                #pragma unroll
                for (int v = 0; v < 8; ++v)
                    xraw[kb][v] = *(const float2*)(xn + a_k0(kb, laneHalf, v));
        }
        if (t + 2 < T_) __builtin_prefetch(xrow + (size_t)(t + 2) * F_, 0, 0);

        // elementwise cell update: 800 items = 16 rows x 50 units
        const bool last = (t == T_ - 1);
        for (int item = tid; item < 16 * H_; item += 128) {
            const int row = item & 15;
            const int j   = item >> 4;
            const float zi = zbuf[(      j) * 16 + row];
            const float zf = zbuf[( 64 + j) * 16 + row];
            const float zg = zbuf[(128 + j) * 16 + row];
            const float zo = zbuf[(192 + j) * 16 + row];
            const float cprev = cbuf[j * 16 + row];
            const float cnew  = sigmoidf_(zf) * cprev + sigmoidf_(zi) * tanhf_(zg);
            const float h     = sigmoidf_(zo) * tanhf_(cnew);
            cbuf[j * 16 + row]   = cnew;
            hbuf[row * KPAD + j] = (_Float16)h;
            if (last) hf[row * H_ + j] = h;
        }
        __syncthreads();   // h,c ready for next step
    }

    // ---- head: out = relu(h@W1 + b1)@W2 + b2 + x[:, -1, 0] ----
    if (tid < 16) {
        rowacc[tid] = b2[0] + x[((size_t)(tile * 16 + tid) * T_ + (T_ - 1)) * F_];
    }
    __syncthreads();
    for (int item = tid; item < 16 * HEAD_; item += 128) {
        const int row = item & 15;
        const int hh  = item >> 4;
        float s = b1[hh];
        #pragma unroll 5
        for (int j = 0; j < H_; ++j) s += hf[row * H_ + j] * W1[j * HEAD_ + hh];
        atomicAdd(&rowacc[row], fmaxf(s, 0.0f) * W2[hh]);   // ds_add_f32
    }
    __syncthreads();
    if (tid < 16) out[tile * 16 + tid] = rowacc[tid];
}

extern "C" void kernel_launch(void* const* d_in, const int* in_sizes, int n_in,
                              void* d_out, int out_size, void* d_ws, size_t ws_size,
                              hipStream_t stream) {
    (void)in_sizes; (void)n_in; (void)d_ws; (void)ws_size; (void)out_size;
    const float* x  = (const float*)d_in[0];
    const float* Wx = (const float*)d_in[1];
    const float* Wh = (const float*)d_in[2];
    const float* b  = (const float*)d_in[3];
    const float* W1 = (const float*)d_in[4];
    const float* b1 = (const float*)d_in[5];
    const float* W2 = (const float*)d_in[6];
    const float* b2 = (const float*)d_in[7];
    float* out = (float*)d_out;

    dim3 grid(B_ / 16);   // 512 blocks of 16 batch rows
    dim3 block(128);      // 4 waves, 4 N-tiles each
    lstm_wmma_kernel<<<grid, block, 0, stream>>>(x, Wx, Wh, b, W1, b1, W2, b2, out);
}
